// ColonyCBF_43224550867069
// MI455X (gfx1250) — matrix operand order repaired and checked by reading, not compiled, not run on hardware
//
#include <hip/hip_runtime.h>
#include <math.h>

#define BATCH      2097152
#define IN_DIM     28
#define HID        64
#define TPW        4            // 16-sample tiles per wave
#define WAVES      8            // waves per block (256 threads, wave32)
#define ROWSTRIDE  144          // bytes per LDS h-row: 64*2 data + 16 pad
#define PSTR       20           // floats per partial-row (80 B, 16B aligned)
#define XBUF       2048         // staging buffer: 1792 B x_all rows + 256 B x_local
#define WLDS       (2*XBUF + 16*ROWSTRIDE)   // 6400 B per wave

typedef __attribute__((ext_vector_type(16))) __bf16  v16bf;
typedef __attribute__((ext_vector_type(16))) float   v16f;
typedef __attribute__((ext_vector_type(8)))  float   v8f;

union BfVec  { v16bf v; uint4 q[2]; };
union FloatV { v16f  v; float4 f4[4]; float e[16]; };

__device__ __forceinline__ unsigned int lds_off(const void* p) {
    // shared->generic cast keeps the LDS byte offset in addr[31:0]
    return (unsigned int)(unsigned long long)p;
}

// Stage one 16-sample tile (x_all rows + x_local rows) into LDS via the
// CDNA5 async-copy path (ASYNCcnt).  5 instructions per tile per wave.
__device__ __forceinline__ void issue_async(const float* __restrict__ x_all,
                                            const float* __restrict__ x_local,
                                            int tile, unsigned int ldsbase,
                                            int lane) {
    const unsigned int g   = (unsigned int)tile * 112u;   // x_all byte offset
    const unsigned int l16 = (unsigned int)lane * 16u;
    asm volatile("global_load_async_to_lds_b128 %0, %1, %2"
                 :: "v"(ldsbase + l16),        "v"(g + l16),        "s"(x_all) : "memory");
    asm volatile("global_load_async_to_lds_b128 %0, %1, %2"
                 :: "v"(ldsbase + 512 + l16),  "v"(g + 512 + l16),  "s"(x_all) : "memory");
    asm volatile("global_load_async_to_lds_b128 %0, %1, %2"
                 :: "v"(ldsbase + 1024 + l16), "v"(g + 1024 + l16), "s"(x_all) : "memory");
    if (lane < 16) {
        asm volatile("global_load_async_to_lds_b128 %0, %1, %2"
                     :: "v"(ldsbase + 1536 + l16), "v"(g + 1536 + l16), "s"(x_all) : "memory");
        asm volatile("global_load_async_to_lds_b128 %0, %1, %2"
                     :: "v"(ldsbase + 1792 + l16),
                        "v"((unsigned int)tile * 16u + l16), "s"(x_local) : "memory");
    }
}

__global__ __launch_bounds__(256) void colony_cbf_kernel(
    const float* __restrict__ x_local, const float* __restrict__ x_all,
    const float* __restrict__ W1, const float* __restrict__ b1,
    const float* __restrict__ W2, const float* __restrict__ b2,
    const float* __restrict__ W3, const float* __restrict__ b3,
    const float* __restrict__ risk_weights, float* __restrict__ out)
{
    __shared__ __align__(16) unsigned char s_lds[WAVES * WLDS];

    const int lane   = threadIdx.x & 31;
    const int ln     = lane & 15;            // column / row-within-half
    const int half   = lane >> 4;            // 0: lanes 0-15, 1: lanes 16-31
    const int waveId = threadIdx.x >> 5;
    unsigned char* wbase = s_lds + waveId * WLDS;
    unsigned char* hrow  = wbase + 2 * XBUF;         // h relayout scratch
    float*         prow  = (float*)hrow;             // layer-3 partials (reused)
    const unsigned int xlds0 = lds_off(wbase);
    const unsigned int xlds1 = xlds0 + XBUF;

    // ---- softmax(|risk_weights|) baseline weights (tiny, per-thread) ----
    const float4 rw = *(const float4*)risk_weights;
    float a0r = fabsf(rw.x), a1r = fabsf(rw.y), a2r = fabsf(rw.z), a3r = fabsf(rw.w);
    float mx  = fmaxf(fmaxf(a0r, a1r), fmaxf(a2r, a3r));
    float e0 = expf(a0r - mx), e1 = expf(a1r - mx), e2 = expf(a2r - mx), e3 = expf(a3r - mx);
    float inv = 1.0f / (e0 + e1 + e2 + e3);
    const float sw0 = e0 * inv, sw1 = e1 * inv, sw2 = e2 * inv, sw3 = e3 * inv;

    // ---- preload weights into WMMA B-layout registers (reused every tile) ----
    // B-layout (16-bit, 32x16): lane ln = column N; lanes 0-15 hold K=0..7,16..23,
    // lanes 16-31 hold K=8..15,24..31; element i = K ((i&7) + half*8 + (i>>3)*16).
    BfVec w1b[4];                 // W1: 28x64, K padded to 32, 4 N-tiles
    #pragma unroll
    for (int nt = 0; nt < 4; ++nt) {
        const int n = nt * 16 + ln;
        FloatV wf;
        #pragma unroll
        for (int i = 0; i < 16; ++i) {
            const int k = (i & 7) + half * 8 + (i >> 3) * 16;
            wf.e[i] = (k < IN_DIM) ? W1[k * HID + n] : 0.f;
        }
        w1b[nt].v = __builtin_convertvector(wf.v, v16bf);
    }
    BfVec w2b[2][2];              // W2: 64x32, 2 K-chunks x 2 N-tiles
    #pragma unroll
    for (int kc = 0; kc < 2; ++kc) {
        #pragma unroll
        for (int nt = 0; nt < 2; ++nt) {
            const int n = nt * 16 + ln;
            FloatV wf;
            #pragma unroll
            for (int i = 0; i < 16; ++i) {
                const int k = kc * 32 + (i & 7) + half * 8 + (i >> 3) * 16;
                wf.e[i] = W2[k * 32 + n];
            }
            w2b[kc][nt].v = __builtin_convertvector(wf.v, v16bf);
        }
    }
    float b1v[4];
    #pragma unroll
    for (int nt = 0; nt < 4; ++nt) b1v[nt] = b1[nt * 16 + ln];
    const float b2v0 = b2[ln], b2v1 = b2[16 + ln];
    const float w3v0 = W3[ln], w3v1 = W3[16 + ln];
    const float b3s  = b3[0];

    const int gw   = blockIdx.x * WAVES + waveId;
    const int base = gw * (TPW * 16);

    // prime the double buffer with tile 0
    issue_async(x_all, x_local, base, xlds0, lane);

    #pragma unroll 1
    for (int t = 0; t < TPW; ++t) {
        const int tile = base + t * 16;
        const unsigned char* xb = wbase + ((t & 1) ? XBUF : 0);

        // kick off next tile's async staging, then wait for current buffer
        if (t + 1 < TPW) {
            issue_async(x_all, x_local, tile + 16,
                        ((t + 1) & 1) ? xlds1 : xlds0, lane);
            asm volatile("s_wait_asynccnt 0x5" ::: "memory");
        } else {
            asm volatile("s_wait_asynccnt 0x0" ::: "memory");
        }

        // ---- build layer-1 A matrix (16x32 bf16, K 28..31 zero-padded) ----
        // x_flat[k] == x_local[s][k] (k<4)  /  x_all row of sample s (k>=4)
        const float* xrow = (const float*)(xb + ln * 112);
        const float* xloc = (const float*)(xb + 1792 + ln * 16);
        FloatV af;
        if (half == 0) {                     // K = 0..7 and 16..23
            af.f4[0] = *(const float4*)(xloc);
            af.f4[1] = *(const float4*)(xrow + 4);
            af.f4[2] = *(const float4*)(xrow + 16);
            af.f4[3] = *(const float4*)(xrow + 20);
        } else {                             // K = 8..15 and 24..31 (28..31 = 0)
            af.f4[0] = *(const float4*)(xrow + 8);
            af.f4[1] = *(const float4*)(xrow + 12);
            af.f4[2] = *(const float4*)(xrow + 24);
            af.f4[3] = make_float4(0.f, 0.f, 0.f, 0.f);
        }
        BfVec a;
        a.v = __builtin_convertvector(af.v, v16bf);

        // ---- layer 1: 16x32 @ 32x64  (4 WMMAs) ----
        v8f c1[4];
        #pragma unroll
        for (int nt = 0; nt < 4; ++nt) {
            v8f z = {};
            c1[nt] = __builtin_amdgcn_wmma_f32_16x16x32_bf16(
                false, a.v, false, w1b[nt].v, (short)0, z, false, false);
        }

        // ---- bias + ReLU, spill to LDS in row-major bf16 ----
        #pragma unroll
        for (int nt = 0; nt < 4; ++nt) {
            const int n = nt * 16 + ln;
            #pragma unroll
            for (int r = 0; r < 8; ++r) {
                const int m = r + half * 8;
                float h = fmaxf(c1[nt][r] + b1v[nt], 0.f);
                ((__bf16*)(hrow + m * ROWSTRIDE))[n] = (__bf16)h;
            }
        }
        asm volatile("s_wait_dscnt 0x0" ::: "memory");

        // ---- reload h as two 16x32 A matrices (A-layout) ----
        const unsigned char* rp = hrow + ln * ROWSTRIDE + half * 16;
        BfVec h0m, h1m;
        h0m.q[0] = *(const uint4*)(rp +  0);
        h0m.q[1] = *(const uint4*)(rp + 32);
        h1m.q[0] = *(const uint4*)(rp + 64);
        h1m.q[1] = *(const uint4*)(rp + 96);

        // ---- layer 2: 16x64 @ 64x32  (2x2 chained WMMAs) ----
        v8f c2[2];
        #pragma unroll
        for (int nt = 0; nt < 2; ++nt) {
            v8f z = {};
            v8f p = __builtin_amdgcn_wmma_f32_16x16x32_bf16(
                false, h0m.v, false, w2b[0][nt].v, (short)0, z, false, false);
            c2[nt] = __builtin_amdgcn_wmma_f32_16x16x32_bf16(
                false, h1m.v, false, w2b[1][nt].v, (short)0, p, false, false);
        }
        asm volatile("s_wait_dscnt 0x0" ::: "memory");

        // ---- layer 3: per-lane partials into LDS (reusing h scratch) ----
        // partial for row m (= r + 8*half) from column n-slice {ln, 16+ln}
        #pragma unroll
        for (int r = 0; r < 8; ++r) {
            float h0v = fmaxf(c2[0][r] + b2v0, 0.f);
            float h1v = fmaxf(c2[1][r] + b2v1, 0.f);
            float acc = fmaf(h0v, w3v0, h1v * w3v1);
            prow[(r + half * 8) * PSTR + ln] = acc;
        }
        asm volatile("s_wait_dscnt 0x0" ::: "memory");

        // ---- sum row ln: each half-wave sums 8 of 16 columns, combine ----
        const float4* pr = (const float4*)(prow + ln * PSTR + half * 8);
        float4 sA = pr[0], sB = pr[1];
        float ssum = ((sA.x + sA.y) + (sA.z + sA.w))
                   + ((sB.x + sB.y) + (sB.z + sB.w));
        ssum += __shfl_xor(ssum, 16, 32);     // lanes 0-15 now hold full rows

        if (half == 0) {
            const int srow = tile + ln;
            float4 xl = *(const float4*)(xb + 1792 + ln * 16);  // from LDS stage
            float lr = sw0 * xl.x + sw1 * xl.y + sw2 * xl.z + sw3 * xl.w;
            out[srow] = (0.3f - lr) + 0.1f * (ssum + b3s);      // coalesced
        }
    }
}

extern "C" void kernel_launch(void* const* d_in, const int* in_sizes, int n_in,
                              void* d_out, int out_size, void* d_ws, size_t ws_size,
                              hipStream_t stream) {
    const float* x_local = (const float*)d_in[0];
    const float* x_all   = (const float*)d_in[1];
    const float* W1      = (const float*)d_in[2];
    const float* b1      = (const float*)d_in[3];
    const float* W2      = (const float*)d_in[4];
    const float* b2      = (const float*)d_in[5];
    const float* W3      = (const float*)d_in[6];
    const float* b3      = (const float*)d_in[7];
    const float* rw      = (const float*)d_in[8];
    float* out           = (float*)d_out;

    const int blocks = BATCH / (16 * WAVES * TPW);   // 4096
    colony_cbf_kernel<<<blocks, 256, 0, stream>>>(
        x_local, x_all, W1, b1, W2, b2, W3, b3, rw, out);
}
